// OmniAttention_6004364280334
// MI455X (gfx1250) — compile-verified, matrix operand order checked by reading, not run on hardware
//
#include <hip/hip_runtime.h>
#include <hip/hip_bf16.h>
#include <cstdint>

typedef __bf16 bf16_t;
typedef __attribute__((ext_vector_type(16))) __bf16 v16bf;
typedef __attribute__((ext_vector_type(8)))  __bf16 v8bf;
typedef __attribute__((ext_vector_type(8)))  float  v8f;
typedef __attribute__((ext_vector_type(4)))  unsigned int u32x4;
typedef __attribute__((ext_vector_type(8)))  int i32x8;
typedef __attribute__((ext_vector_type(4)))  int i32x4;

#define WMMA_BF16(a, b, c) \
  __builtin_amdgcn_wmma_f32_16x16x32_bf16(false, (a), false, (b), (short)0, (c), false, false)

#if __has_builtin(__builtin_amdgcn_tensor_load_to_lds)
#define USE_TDM 1
#else
#define USE_TDM 0
#endif

// Problem dims
constexpr int Bb = 2, Nn = 2048, Dd = 1024, Hh = 16, HD = 64;
constexpr float ATT_SCALE = 0.125f;  // HD^-0.5

// Workspace layout (bf16 element offsets)
constexpr size_t OFF_XBF   = 0;          // [4096,1024]
constexpr size_t OFF_WQKV  = 4194304;    // [3072,1024]
constexpr size_t OFF_WPROJ = 7340032;    // [1024,1024]
constexpr size_t OFF_Q     = 8388608;    // [B,H,N,HD]
constexpr size_t OFF_K     = 12582912;   // [B,H,N,HD]
constexpr size_t OFF_VT    = 16777216;   // [B,H,HD,N]
constexpr size_t OFF_AO    = 20971520;   // [4096,1024]

// ---------- helpers ----------

__global__ void cvt_bf16(const float* __restrict__ src, bf16_t* __restrict__ dst, int n) {
  int i = blockIdx.x * blockDim.x + threadIdx.x;
  int stride = gridDim.x * blockDim.x;
  for (; i < n; i += stride) dst[i] = (bf16_t)src[i];
}

// 16x32 bf16 fragment in the ISA A/B striping: lane L holds row L%16;
// VGPR0..3 = K halves (L<16 ? 0..7 : 8..15), VGPR4..7 = same +16.
__device__ inline v16bf load_frag(const bf16_t* __restrict__ base, int ld,
                                  int row0, int k0, int lane) {
  const int r  = row0 + (lane & 15);
  const int ko = k0 + ((lane >> 4) << 3);
  const bf16_t* p = base + (size_t)r * ld + ko;
  v8bf lo = *(const v8bf*)p;
  v8bf hi = *(const v8bf*)(p + 16);
  v16bf f;
#pragma unroll
  for (int i = 0; i < 8; ++i) { f[i] = lo[i]; f[i + 8] = hi[i]; }
  return f;
}

#if USE_TDM
// ISA flat-aperture rule: LDS generic VA bits [31:0] are the LDS byte offset.
__device__ inline uint32_t lds_addr_of(const void* p) {
  return (uint32_t)(uintptr_t)p;
}

// Issue one 2D TDM tile load (bf16 elements) per CDNA5 D# layout (ISA 08 §8).
// dim0/stride0/dim1 describe the tensor; tile0 x tile1 the tile at gptr.
__device__ inline void tdm_load_2d(uint32_t lds_byte, const void* gptr,
                                   uint32_t dim0, uint32_t stride0, uint32_t dim1,
                                   uint32_t tile0, uint32_t tile1) {
  uint64_t ga = (uint64_t)(uintptr_t)gptr;
  u32x4 g0;
  g0[0] = 1u;                                                  // count=1 (valid), user mode
  g0[1] = lds_byte;                                            // lds_addr
  g0[2] = (uint32_t)(ga & 0xFFFFFFFFu);                        // global_addr[31:0]
  g0[3] = (uint32_t)((ga >> 32) & 0x01FFFFFFu) | (2u << 30);   // global_addr[56:32], type=2
  i32x8 g1;
  g1[0] = (int)(1u << 16);                                     // data_size=1 (2 bytes)
  g1[1] = (int)((dim0 & 0xFFFFu) << 16);                       // tensor_dim0[15:0]
  g1[2] = (int)((dim0 >> 16) | ((dim1 & 0xFFFFu) << 16));      // dim0[31:16], dim1[15:0]
  g1[3] = (int)((dim1 >> 16) | (tile0 << 16));                 // dim1[31:16], tile_dim0
  g1[4] = (int)(tile1);                                        // tile_dim1 (tile_dim2=0)
  g1[5] = (int)(stride0);                                      // tensor_dim0_stride[31:0]
  g1[6] = 0;                                                   // stride0[47:32], stride1 lo
  g1[7] = 0;
  i32x4 z4 = {0, 0, 0, 0};
#if defined(__clang_major__) && __clang_major__ >= 23
  i32x8 z8 = {0, 0, 0, 0, 0, 0, 0, 0};
  __builtin_amdgcn_tensor_load_to_lds(g0, g1, z4, z4, z8, 0);
#else
  __builtin_amdgcn_tensor_load_to_lds(g0, g1, z4, z4, 0);
#endif
}
#endif  // USE_TDM

// ---------- kernel 1: QKV projection GEMM + scatter ----------
__global__ __launch_bounds__(256)
void gemm_qkv(const bf16_t* __restrict__ xbf, const bf16_t* __restrict__ wbf,
              const float* __restrict__ bias,
              bf16_t* __restrict__ qo, bf16_t* __restrict__ ko,
              bf16_t* __restrict__ vto) {
  const int lane = threadIdx.x & 31;
  const int wave = threadIdx.x >> 5;
  const int n0 = blockIdx.x * 64 + (wave & 1) * 32;
  const int m0 = blockIdx.y * 128 + (wave >> 1) * 32;

  v8f c[2][2] = {};
  for (int kk = 0; kk < Dd; kk += 32) {
    if (kk + 32 < Dd) {
      __builtin_prefetch(xbf + (size_t)(m0 + (lane & 15)) * Dd + kk + 32, 0, 3);
      __builtin_prefetch(wbf + (size_t)(n0 + (lane & 15)) * Dd + kk + 32, 0, 3);
    }
    v16bf a0 = load_frag(xbf, Dd, m0,      kk, lane);
    v16bf a1 = load_frag(xbf, Dd, m0 + 16, kk, lane);
    v16bf b0 = load_frag(wbf, Dd, n0,      kk, lane);
    v16bf b1 = load_frag(wbf, Dd, n0 + 16, kk, lane);
    c[0][0] = WMMA_BF16(a0, b0, c[0][0]);
    c[0][1] = WMMA_BF16(a0, b1, c[0][1]);
    c[1][0] = WMMA_BF16(a1, b0, c[1][0]);
    c[1][1] = WMMA_BF16(a1, b1, c[1][1]);
  }

  const int nlo = lane & 15;
  const int mhi = (lane >> 4) * 8;
#pragma unroll
  for (int ti = 0; ti < 2; ++ti)
#pragma unroll
    for (int tj = 0; tj < 2; ++tj)
#pragma unroll
      for (int i = 0; i < 8; ++i) {
        int mg = m0 + ti * 16 + mhi + i;
        int e  = n0 + tj * 16 + nlo;
        float v = c[ti][tj][i] + bias[e];
        int b = mg >> 11, n = mg & 2047;
        int sect = e >> 10, r = e & 1023;
        int h = r >> 6, d = r & 63;
        bf16_t bv = (bf16_t)v;
        if (sect == 0)
          qo[(((size_t)(b * Hh + h)) * Nn + n) * HD + d] = bv;
        else if (sect == 1)
          ko[(((size_t)(b * Hh + h)) * Nn + n) * HD + d] = bv;
        else
          vto[(((size_t)(b * Hh + h)) * HD + d) * Nn + n] = bv;
      }
}

// ---------- kernel 2: flash attention with TDM-staged K/V tiles ----------
// Block = 4 waves sharing one (b,h); K (32x64) and Vt (64x32) tiles are
// DMA'd into LDS once per block (double-buffered), WMMA fragments read LDS.
__global__ __launch_bounds__(128)
void flash_attn(const bf16_t* __restrict__ qbf, const bf16_t* __restrict__ kbf,
                const bf16_t* __restrict__ vtbf, bf16_t* __restrict__ aout) {
  __shared__ bf16_t ktile[2][32 * 64];   // [key][d]
  __shared__ bf16_t vtile[2][64 * 32];   // [d][key]
  __shared__ bf16_t pshare[4][16 * 32];

  const int lane = threadIdx.x & 31;
  const int wave = threadIdx.x >> 5;
  const int bh = blockIdx.y;                 // b*16+h
  const int q0 = blockIdx.x * 64 + wave * 16;

  const bf16_t* qb = qbf + (size_t)bh * Nn * HD;
  const bf16_t* kb = kbf + (size_t)bh * Nn * HD;
  const bf16_t* vb = vtbf + (size_t)bh * HD * Nn;

  v16bf aq0 = load_frag(qb, HD, q0, 0, lane);
  v16bf aq1 = load_frag(qb, HD, q0, 32, lane);

  v8f oacc[4] = {};
  float mst[8], lst[8];
#pragma unroll
  for (int i = 0; i < 8; ++i) { mst[i] = -__builtin_inff(); lst[i] = 0.f; }

  bf16_t* pw = &pshare[wave][0];
  const int nlo = lane & 15;
  const int mhi = (lane >> 4) * 8;

#if USE_TDM
  // Prologue: DMA tile 0 into buffer 0.
  if (wave == 0) {
    tdm_load_2d(lds_addr_of(&ktile[0][0]), kb, HD, HD, Nn, HD, 32);
    tdm_load_2d(lds_addr_of(&vtile[0][0]), vb, Nn, Nn, HD, 32, HD);
  }
#endif

  for (int key0 = 0; key0 < Nn; key0 += 32) {
    const int cur = (key0 >> 5) & 1;
#if USE_TDM
    if (wave == 0) {
      if (key0 + 32 < Nn) {
        // Kick next tile DMA into the other buffer, then wait for current
        // pair (in-order TENSORcnt: <=2 means the first pair landed).
        tdm_load_2d(lds_addr_of(&ktile[cur ^ 1][0]),
                    kb + (size_t)(key0 + 32) * HD, HD, HD, Nn, HD, 32);
        tdm_load_2d(lds_addr_of(&vtile[cur ^ 1][0]),
                    vb + (key0 + 32), Nn, Nn, HD, 32, HD);
        __builtin_amdgcn_s_wait_tensorcnt(2);
      } else {
        __builtin_amdgcn_s_wait_tensorcnt(0);
      }
    }
    __syncthreads();  // current buffer visible to all waves
#else
    // Fallback: cooperative vector copy of current tile (single buffer).
    __syncthreads();
    {
      const int t = threadIdx.x;
      v8bf* kdst = (v8bf*)&ktile[0][0];
      const v8bf* ksrc = (const v8bf*)(kb + (size_t)key0 * HD);
      kdst[t * 2]     = ksrc[t * 2];
      kdst[t * 2 + 1] = ksrc[t * 2 + 1];
      v8bf* vdst = (v8bf*)&vtile[0][0];
#pragma unroll
      for (int j = 0; j < 2; ++j) {
        int idx = t * 2 + j;            // 256 v8bf chunks: row=idx>>2, col8=idx&3
        vdst[idx] = *(const v8bf*)(vb + (size_t)(idx >> 2) * Nn + key0 + (idx & 3) * 8);
      }
    }
    __syncthreads();
#endif
    const bf16_t* kt = &ktile[USE_TDM ? cur : 0][0];
    const bf16_t* vt = &vtile[USE_TDM ? cur : 0][0];

    // S = Q Kt over HD=64 contraction (fragments from LDS: ds_load_b128)
    v8f s0 = {}, s1 = {};
    {
      v16bf bk00 = load_frag(kt, HD, 0,  0,  lane);
      v16bf bk10 = load_frag(kt, HD, 0,  32, lane);
      v16bf bk01 = load_frag(kt, HD, 16, 0,  lane);
      v16bf bk11 = load_frag(kt, HD, 16, 32, lane);
      s0 = WMMA_BF16(aq0, bk00, s0);
      s0 = WMMA_BF16(aq1, bk10, s0);
      s1 = WMMA_BF16(aq0, bk01, s1);
      s1 = WMMA_BF16(aq1, bk11, s1);
    }

#pragma unroll
    for (int i = 0; i < 8; ++i) {
      float v0 = s0[i] * ATT_SCALE, v1 = s1[i] * ATT_SCALE;
      float tmax = fmaxf(v0, v1);
#pragma unroll
      for (int msk = 8; msk >= 1; msk >>= 1)
        tmax = fmaxf(tmax, __shfl_xor(tmax, msk, 32));
      float nm = fmaxf(mst[i], tmax);
      float alpha = __expf(mst[i] - nm);
      float p0 = __expf(v0 - nm), p1 = __expf(v1 - nm);
      float rs = p0 + p1;
#pragma unroll
      for (int msk = 8; msk >= 1; msk >>= 1) rs += __shfl_xor(rs, msk, 32);
      lst[i] = lst[i] * alpha + rs;
      mst[i] = nm;
#pragma unroll
      for (int t = 0; t < 4; ++t) oacc[t][i] *= alpha;
      int m = mhi + i;
      pw[m * 32 + nlo]      = (bf16_t)p0;
      pw[m * 32 + 16 + nlo] = (bf16_t)p1;
    }
    // wave-local LDS RAW hazard (per-wave region): stores -> loads
    asm volatile("s_wait_dscnt 0" ::: "memory");

    v16bf pa;
    {
      const bf16_t* pp = pw + (lane & 15) * 32 + ((lane >> 4) << 3);
      v8bf lo = *(const v8bf*)pp;
      v8bf hi = *(const v8bf*)(pp + 16);
#pragma unroll
      for (int i = 0; i < 8; ++i) { pa[i] = lo[i]; pa[i + 8] = hi[i]; }
    }

    // O += P @ V (contraction = 32 keys; Vt rows are d)
#pragma unroll
    for (int t = 0; t < 4; ++t) {
      v16bf bv = load_frag(vt, 32, t * 16, 0, lane);
      oacc[t] = WMMA_BF16(pa, bv, oacc[t]);
    }
    __syncthreads();  // all waves done reading before buffer reuse
  }

  const int h = bh & 15;
  const int b = bh >> 4;
#pragma unroll
  for (int i = 0; i < 8; ++i) {
    float inv = 1.0f / lst[i];
    int qrow = q0 + mhi + i;
    size_t rowoff = ((size_t)b * Nn + qrow) * Dd + h * HD;
#pragma unroll
    for (int t = 0; t < 4; ++t)
      aout[rowoff + t * 16 + nlo] = (bf16_t)(oacc[t][i] * inv);
  }
}

// ---------- kernel 3: output projection GEMM ----------
__global__ __launch_bounds__(256)
void gemm_proj(const bf16_t* __restrict__ abf, const bf16_t* __restrict__ wbf,
               const float* __restrict__ bias, float* __restrict__ out) {
  const int lane = threadIdx.x & 31;
  const int wave = threadIdx.x >> 5;
  const int n0 = blockIdx.x * 64 + (wave & 1) * 32;
  const int m0 = blockIdx.y * 128 + (wave >> 1) * 32;

  v8f c[2][2] = {};
  for (int kk = 0; kk < Dd; kk += 32) {
    if (kk + 32 < Dd) {
      __builtin_prefetch(abf + (size_t)(m0 + (lane & 15)) * Dd + kk + 32, 0, 3);
      __builtin_prefetch(wbf + (size_t)(n0 + (lane & 15)) * Dd + kk + 32, 0, 3);
    }
    v16bf a0 = load_frag(abf, Dd, m0,      kk, lane);
    v16bf a1 = load_frag(abf, Dd, m0 + 16, kk, lane);
    v16bf b0 = load_frag(wbf, Dd, n0,      kk, lane);
    v16bf b1 = load_frag(wbf, Dd, n0 + 16, kk, lane);
    c[0][0] = WMMA_BF16(a0, b0, c[0][0]);
    c[0][1] = WMMA_BF16(a0, b1, c[0][1]);
    c[1][0] = WMMA_BF16(a1, b0, c[1][0]);
    c[1][1] = WMMA_BF16(a1, b1, c[1][1]);
  }

  const int nlo = lane & 15;
  const int mhi = (lane >> 4) * 8;
#pragma unroll
  for (int ti = 0; ti < 2; ++ti)
#pragma unroll
    for (int tj = 0; tj < 2; ++tj)
#pragma unroll
      for (int i = 0; i < 8; ++i) {
        int mg = m0 + ti * 16 + mhi + i;
        int e  = n0 + tj * 16 + nlo;
        out[(size_t)mg * Dd + e] = c[ti][tj][i] + bias[e];
      }
}

// ---------- launcher ----------
extern "C" void kernel_launch(void* const* d_in, const int* in_sizes, int n_in,
                              void* d_out, int out_size, void* d_ws, size_t ws_size,
                              hipStream_t stream) {
  (void)in_sizes; (void)n_in; (void)out_size; (void)ws_size;
  const float* x      = (const float*)d_in[0];
  const float* w_qkv  = (const float*)d_in[1];
  const float* b_qkv  = (const float*)d_in[2];
  const float* w_proj = (const float*)d_in[3];
  const float* b_proj = (const float*)d_in[4];
  float* out = (float*)d_out;

  bf16_t* ws      = (bf16_t*)d_ws;
  bf16_t* xbf     = ws + OFF_XBF;
  bf16_t* wqkvbf  = ws + OFF_WQKV;
  bf16_t* wprojbf = ws + OFF_WPROJ;
  bf16_t* qbf     = ws + OFF_Q;
  bf16_t* kbf     = ws + OFF_K;
  bf16_t* vtbf    = ws + OFF_VT;
  bf16_t* aobf    = ws + OFF_AO;

  cvt_bf16<<<1024, 256, 0, stream>>>(x,      xbf,     Bb * Nn * Dd);
  cvt_bf16<<<1024, 256, 0, stream>>>(w_qkv,  wqkvbf,  3 * Dd * Dd);
  cvt_bf16<<<512,  256, 0, stream>>>(w_proj, wprojbf, Dd * Dd);

  gemm_qkv<<<dim3(48, 32), 256, 0, stream>>>(xbf, wqkvbf, b_qkv, qbf, kbf, vtbf);
  flash_attn<<<dim3(Nn / 64, Bb * Hh), 128, 0, stream>>>(qbf, kbf, vtbf, aobf);
  gemm_proj<<<dim3(16, 32), 256, 0, stream>>>(aobf, wprojbf, b_proj, out);
}
